// Simplified_SelfAttention_26362509262867
// MI455X (gfx1250) — matrix operand order repaired and checked by reading, not compiled, not run on hardware
//
#include <hip/hip_runtime.h>

typedef __bf16 bf16;
typedef bf16 v16bf __attribute__((ext_vector_type(16)));
typedef bf16 v8bf  __attribute__((ext_vector_type(8)));
typedef float v8f  __attribute__((ext_vector_type(8)));
typedef int  v4i   __attribute__((ext_vector_type(4)));

#define N_    64
#define P_    1024
#define E_    512
#define H_    8
#define R_    64
#define AEXT_ 2112   // 2*P + 64 replicated alpha for circulant indexing

// ---- CDNA5 async global->LDS copy (ASYNCcnt path), with safe fallbacks ----
#if __has_builtin(__builtin_amdgcn_global_load_async_to_lds_b128)
#define USE_ASYNC_LDS 1
typedef __attribute__((address_space(1))) v4i as1_v4i;   // global (prints as __device__)
typedef __attribute__((address_space(3))) v4i as3_v4i;   // LDS    (prints as __shared__)
__device__ __forceinline__ void async_b128(const bf16* g, bf16* l) {
  __builtin_amdgcn_global_load_async_to_lds_b128((as1_v4i*)g, (as3_v4i*)l, 0, 0);
}
__device__ __forceinline__ void wait_async_le(int n) {
#if __has_builtin(__builtin_amdgcn_s_wait_asynccnt)
  if (n == 0)      __builtin_amdgcn_s_wait_asynccnt(0);
  else             __builtin_amdgcn_s_wait_asynccnt(3);
#else
  if (n == 0) asm volatile("s_wait_asynccnt 0" ::: "memory");
  else        asm volatile("s_wait_asynccnt 3" ::: "memory");
#endif
}
#else
#define USE_ASYNC_LDS 0
#endif

// ---- unified NT-form WMMA GEMM core: C[m,n] = sum_k A[m,k]*B[n,k] ----
static constexpr int BM = 128, BN = 64, BK = 32, PADK = 8;
static constexpr int LDSA = BK + PADK, LDSB = BK + PADK, NT = 256;

__device__ __forceinline__ v16bf frag_a_lds(const bf16* As, int mOff, int lo, int hi) {
  // A 16x32 bf16: lane holds row m, 16 consecutive K starting at hi*16
  const bf16* row = As + (mOff + lo) * LDSA;
  union { v16bf v; v8bf h[2]; } u;
  u.h[0] = *(const v8bf*)(row + hi * 16);
  u.h[1] = *(const v8bf*)(row + hi * 16 + 8);
  return u.v;
}

__device__ __forceinline__ v16bf frag_b_lds(const bf16* Bs, int nOff, int lo, int hi) {
  // B 32x16 bf16: lane holds col n, 16 consecutive K starting at hi*16
  const bf16* row = Bs + (nOff + lo) * LDSB;
  union { v16bf v; v8bf h[2]; } u;
  u.h[0] = *(const v8bf*)(row + hi * 16);
  u.h[1] = *(const v8bf*)(row + hi * 16 + 8);
  return u.v;
}

// EPI: 0 = store bf16, 1 = +bias, log_cosh, store f32.  AALIGNED: A rows 16B-loadable.
template<int EPI, bool AALIGNED>
__device__ __forceinline__ void gemm_core(const bf16* __restrict__ Ag, long lda,
                                          const bf16* __restrict__ Bg, long ldb,
                                          void* __restrict__ Cg, long ldc,
                                          const float* __restrict__ bias, int K)
{
  __shared__ bf16 As[2 * BM * LDSA];   // double buffered: 2 x 10 KB
  __shared__ bf16 Bs[2 * BN * LDSB];   // double buffered: 2 x  5 KB

  const int tid  = threadIdx.x;
  const int lane = tid & 31;
  const int wave = tid >> 5;
  const int lo   = lane & 15;
  const int hi   = lane >> 4;
  const int wm   = (wave & 3) * 32;   // 4 waves along M -> 128
  const int wn   = (wave >> 2) * 32;  // 2 waves along N -> 64

  v8f acc[2][2] = {};

  auto stage_tiles = [&](int buf, int kb) {
    bf16* Asb = As + buf * (BM * LDSA);
    bf16* Bsb = Bs + buf * (BN * LDSB);
    if (AALIGNED) {
      #pragma unroll
      for (int it = 0; it < (BM * (BK / 8)) / NT; ++it) {   // 2 chunks/thread
        int i = tid + it * NT;
        int r = i >> 2, c8 = (i & 3) << 3;
#if USE_ASYNC_LDS
        async_b128(Ag + (long)r * lda + kb + c8, Asb + r * LDSA + c8);
#else
        *(v8bf*)(Asb + r * LDSA + c8) = *(const v8bf*)(Ag + (long)r * lda + kb + c8);
#endif
      }
    } else {
      #pragma unroll
      for (int it = 0; it < (BM * BK) / NT; ++it) {         // tiny cache-hot alpha
        int i = tid + it * NT;
        int r = i >> 5, c = i & 31;
        Asb[r * LDSA + c] = Ag[(long)r * lda + kb + c];
      }
    }
    {
      int r = tid >> 2, c8 = (tid & 3) << 3;                // 1 chunk/thread
#if USE_ASYNC_LDS
      async_b128(Bg + (long)r * ldb + kb + c8, Bsb + r * LDSB + c8);
#else
      *(v8bf*)(Bsb + r * LDSB + c8) = *(const v8bf*)(Bg + (long)r * ldb + kb + c8);
      __builtin_prefetch(Bg + (long)r * ldb + kb + BK + c8, 0, 1);
#endif
    }
  };

  stage_tiles(0, 0);
  int cur = 0;

  for (int kb = 0; kb < K; kb += BK) {
    const bool hasNext = (kb + BK) < K;
    if (hasNext) stage_tiles(cur ^ 1, kb + BK);   // overlap next fill with compute
#if USE_ASYNC_LDS
    wait_async_le(hasNext ? 3 : 0);               // current buffer's fills done (in-order)
#endif
    __syncthreads();                              // all waves' fills visible

    const bf16* Asb = As + cur * (BM * LDSA);
    const bf16* Bsb = Bs + cur * (BN * LDSB);
    v16bf a0 = frag_a_lds(Asb, wm,      lo, hi);
    v16bf a1 = frag_a_lds(Asb, wm + 16, lo, hi);
    v16bf b0 = frag_b_lds(Bsb, wn,      lo, hi);
    v16bf b1 = frag_b_lds(Bsb, wn + 16, lo, hi);

    acc[0][0] = __builtin_amdgcn_wmma_f32_16x16x32_bf16(false, a0, false, b0, (short)0, acc[0][0], false, false);
    acc[0][1] = __builtin_amdgcn_wmma_f32_16x16x32_bf16(false, a0, false, b1, (short)0, acc[0][1], false, false);
    acc[1][0] = __builtin_amdgcn_wmma_f32_16x16x32_bf16(false, a1, false, b0, (short)0, acc[1][0], false, false);
    acc[1][1] = __builtin_amdgcn_wmma_f32_16x16x32_bf16(false, a1, false, b1, (short)0, acc[1][1], false, false);

    __syncthreads();                              // done reading 'cur'; next iter may refill it
    cur ^= 1;
  }

  // ---- epilogue: C VGPR v -> row = v + hi*8, col = lo ----
  #pragma unroll
  for (int i = 0; i < 2; ++i)
    #pragma unroll
    for (int j = 0; j < 2; ++j)
      #pragma unroll
      for (int v = 0; v < 8; ++v) {
        int row = wm + i * 16 + hi * 8 + v;
        int col = wn + j * 16 + lo;
        float t = acc[i][j][v];
        if (EPI == 0) {
          ((bf16*)Cg)[(long)row * ldc + col] = (bf16)t;
        } else {
          t += bias[col];
          float a = fabsf(t);
          float r = a + log1pf(__expf(-2.0f * a)) - 0.69314718055994530942f;
          ((float*)Cg)[(long)row * ldc + col] = r;
        }
      }
}

// ---- conversion kernels ----
__global__ __launch_bounds__(256) void k_convert_x(const float* __restrict__ x,
                                                   bf16* __restrict__ xb, long n) {
  long i = (long)blockIdx.x * blockDim.x + threadIdx.x;
  long stride = (long)gridDim.x * blockDim.x;
  for (; i < n; i += stride) xb[i] = (bf16)x[i];
}

__global__ __launch_bounds__(256) void k_convert_weights(const float* __restrict__ V,
                                                         const float* __restrict__ W,
                                                         const float* __restrict__ alpha,
                                                         bf16* __restrict__ vb,
                                                         bf16* __restrict__ wtb,
                                                         bf16* __restrict__ aext) {
  int i = blockIdx.x * 256 + threadIdx.x;          // grid covers E*E
  vb[i] = (bf16)V[i];                              // V flat [h*R+r][d] already K-contiguous
  int eo = i >> 9, ei = i & (E_ - 1);
  wtb[i] = (bf16)W[ei * E_ + eo];                  // Wt[eo][ei] = W[ei][eo]
  if (i < H_ * AEXT_) {
    int h = i / AEXT_, t = i % AEXT_;
    aext[i] = (bf16)alpha[h * P_ + (t & (P_ - 1))]; // circulant replication
  }
}

// ---- stage 1: ws_t[n][c][p] = sum_d Vf[c,d] * x[n,p,d]   (M=E, N=P, K=E) ----
__global__ __launch_bounds__(256) void k_stage1(const bf16* __restrict__ vb,
                                                const bf16* __restrict__ xb,
                                                bf16* __restrict__ wst) {
  int bx = blockIdx.x;
  int n = bx >> 6, r = bx & 63, mb = r >> 4, nb = r & 15;   // 4 m-blocks x 16 n-blocks
  const bf16* Ag = vb + (long)mb * BM * E_;
  const bf16* Bg = xb + (long)n * P_ * E_ + (long)nb * BN * E_;
  bf16* Cg = wst + (long)n * E_ * P_ + (long)mb * BM * P_ + nb * BN;
  gemm_core<0, true>(Ag, E_, Bg, E_, Cg, P_, nullptr, E_);
}

// ---- stage 2: y[n][p][h*R+r] = sum_j alpha[h,(j-p)%P] * ws_t[n][h*R+r][j]  (M=P, N=R, K=P) ----
__global__ __launch_bounds__(256) void k_stage2(const bf16* __restrict__ aext,
                                                const bf16* __restrict__ wst,
                                                bf16* __restrict__ yb) {
  int bx = blockIdx.x;
  int n = bx >> 6, r = bx & 63, h = r >> 3, mb = r & 7;     // 8 heads x 8 m-blocks
  const bf16* Ag = aext + h * AEXT_ + P_ - mb * BM;         // circulant: lda = -1
  const bf16* Bg = wst + (long)n * E_ * P_ + (long)h * R_ * P_;
  bf16* Cg = yb + (long)n * P_ * E_ + (long)mb * BM * E_ + h * R_;
  gemm_core<0, false>(Ag, -1, Bg, P_, Cg, E_, nullptr, P_);
}

// ---- stage 3: out = log_cosh(y @ W + b)   (M=P, N=E, K=E) ----
__global__ __launch_bounds__(256) void k_stage3(const bf16* __restrict__ yb,
                                                const bf16* __restrict__ wtb,
                                                const float* __restrict__ b,
                                                float* __restrict__ out) {
  int bx = blockIdx.x;
  int n = bx >> 6, r = bx & 63, mb = r >> 3, nb = r & 7;    // 8 m-blocks x 8 n-blocks
  const bf16* Ag = yb + (long)n * P_ * E_ + (long)mb * BM * E_;
  const bf16* Bg = wtb + (long)nb * BN * E_;
  float* Cg = out + (long)n * P_ * E_ + (long)mb * BM * E_ + nb * BN;
  gemm_core<1, true>(Ag, E_, Bg, E_, Cg, E_, b + nb * BN, E_);
}

extern "C" void kernel_launch(void* const* d_in, const int* in_sizes, int n_in,
                              void* d_out, int out_size, void* d_ws, size_t ws_size,
                              hipStream_t stream) {
  const float* x     = (const float*)d_in[0];
  const float* alpha = (const float*)d_in[1];
  const float* V     = (const float*)d_in[2];
  const float* W     = (const float*)d_in[3];
  const float* b     = (const float*)d_in[4];

  char* ws = (char*)d_ws;
  const size_t XB = (size_t)N_ * P_ * E_ * sizeof(bf16);   // 64 MiB per big buffer
  bf16* xb   = (bf16*)(ws);
  bf16* wst  = (bf16*)(ws + XB);
  bf16* yb   = (bf16*)(ws + 2 * XB);
  bf16* vb   = (bf16*)(ws + 3 * XB);
  bf16* wtb  = (bf16*)(ws + 3 * XB + (size_t)E_ * E_ * sizeof(bf16));
  bf16* aext = (bf16*)(ws + 3 * XB + (size_t)E_ * E_ * 2 * sizeof(bf16));

  const long totalX = (long)N_ * P_ * E_;
  k_convert_x<<<2048, 256, 0, stream>>>(x, xb, totalX);
  k_convert_weights<<<(E_ * E_) / 256, 256, 0, stream>>>(V, W, alpha, vb, wtb, aext);
  k_stage1<<<4096, 256, 0, stream>>>(vb, xb, wst);
  k_stage2<<<4096, 256, 0, stream>>>(aext, wst, yb);
  k_stage3<<<4096, 256, 0, stream>>>(yb, wtb, b, (float*)d_out);
}
